// FeedbackMetaModelLstmAttention_74311524155781
// MI455X (gfx1250) — matrix operand order repaired and checked by reading, not compiled, not run on hardware
//
#include <hip/hip_runtime.h>
#include <hip/hip_bf16.h>

// ---------------------------------------------------------------------------
// FeedbackMetaModelLstmAttention for MI455X (gfx1250, wave32, WMMA).
// B=32 L=512 F=768 H=512 NH=16 HD=32 HC=256 NL=3, MAXPOS=512.
// All dense GEMMs run on v_wmma_f32_16x16x32_f16 (f16 A/B, f32 accum) with
// 32x64 register blocking per wave (8 WMMAs / K-step, ~21 FLOP/B from cache).
// Attention is fully fused (flash-style) incl. the relative-position term,
// which is computed with WMMA against a 48-row band of rel_emb per tile pair.
// ---------------------------------------------------------------------------

typedef __attribute__((ext_vector_type(16))) _Float16 v16h;
typedef __attribute__((ext_vector_type(8)))  float    v8f;

#define BB  32
#define LL  512
#define FF  768
#define HH  512
#define NH  16
#define HD  32
#define HC  256
#define NLB 3

// ---------------- WMMA fragment helpers (wave32 layouts, ISA 7.12.2) -------

// 16-bit A 16x32, contiguous k order used here (not ISA table order): lanes 0-15 -> M=lane, K=0..15;
// lanes 16-31 -> M=lane-16, K=16..31. Pairs contiguous.
__device__ __forceinline__ v16h load_a_frag(const _Float16* A, int lda,
                                            int m0, int k0) {
  int lane = threadIdx.x & 31;
  int half = lane >> 4;
  int m = m0 + (lane & 15);
  const _Float16* row = A + (size_t)m * lda;
  union { v16h v; unsigned int u[8]; } r;
#pragma unroll
  for (int j = 0; j < 8; ++j) {
    int k = k0 + 2 * j + half * 16;
    r.u[j] = *(const unsigned int*)(row + k);
  }
  return r.v;
}

// 16-bit B-matrix KxN=32x16 from N-major storage (weight rows are K-contiguous):
// lane n = lane&15 holds column n; lanes 0-15 K=0..15, lanes 16-31 K=16..31.
__device__ __forceinline__ v16h load_b_frag_nk(const _Float16* Bt, int ldb,
                                               int n0, int k0) {
  int lane = threadIdx.x & 31;
  int n = n0 + (lane & 15);
  int koff = k0 + ((lane >> 4) << 4);
  const _Float16* row = Bt + (size_t)n * ldb;
  union { v16h v; unsigned int u[8]; } r;
#pragma unroll
  for (int j = 0; j < 8; ++j)
    r.u[j] = *(const unsigned int*)(row + koff + 2 * j);
  return r.v;
}

// 16-bit B-matrix 32x16 from K-major storage (e.g. V: [m][d]).
__device__ __forceinline__ v16h load_b_frag_kn(const _Float16* Bm, int ldb,
                                               int k0, int n0) {
  int lane = threadIdx.x & 31;
  int n = n0 + (lane & 15);
  int koff = k0 + ((lane >> 4) << 4);
  union { v16h v; _Float16 h[16]; } r;
#pragma unroll
  for (int j = 0; j < 16; ++j)
    r.h[j] = Bm[(size_t)(koff + j) * ldb + n];
  return r.v;
}

__device__ __forceinline__ v8f wmma_f16(v16h a, v16h b, v8f c) {
  return __builtin_amdgcn_wmma_f32_16x16x32_f16(false, a, false, b,
                                                (short)0, c, false, false);
}

// ---------------- elementwise f32 -> f16 (with zero padding) ---------------

__global__ void cvt_f16_kernel(const float* __restrict__ src,
                               _Float16* __restrict__ dst, int n, int srcn) {
  int i = blockIdx.x * blockDim.x + threadIdx.x;
  if (i < n) dst[i] = (i < srcn) ? (_Float16)src[i] : (_Float16)0.0f;
}

// ---------------- generic WMMA GEMM: C[M,N] = A[M,K] * W[N,K]^T + bias -----
// 4 waves / block; each wave owns a 32x64 tile: 2 A-frags x 4 B-frags,
// 8 v_wmma per K-step. Block tile = 128(M) x 64(N).

template <typename OutT>
__global__ void gemm_wmma_kernel(const _Float16* __restrict__ A,
                                 const _Float16* __restrict__ Wt,
                                 const float* __restrict__ bias,
                                 OutT* __restrict__ C,
                                 int M, int N, int K,
                                 int lda, int ldb, int ldc) {
  int wave = threadIdx.x >> 5;
  int lane = threadIdx.x & 31;
  int n0 = blockIdx.x * 64;
  int m0 = blockIdx.y * 128 + wave * 32;
  v8f acc0[4] = {};
  v8f acc1[4] = {};
#pragma unroll 2
  for (int k0 = 0; k0 < K; k0 += 32) {
    v16h a0 = load_a_frag(A, lda, m0, k0);
    v16h a1 = load_a_frag(A, lda, m0 + 16, k0);
    if (k0 + 32 < K) { // pull next K-slab toward the caches (global_prefetch_b8)
      __builtin_prefetch(A + (size_t)(m0 + (lane & 15)) * lda + k0 + 32, 0, 3);
      __builtin_prefetch(Wt + (size_t)(n0 + (lane & 31)) * ldb + k0 + 32, 0, 3);
    }
#pragma unroll
    for (int j = 0; j < 4; ++j) {
      v16h b = load_b_frag_nk(Wt, ldb, n0 + 16 * j, k0);
      acc0[j] = wmma_f16(a0, b, acc0[j]);
      acc1[j] = wmma_f16(a1, b, acc1[j]);
    }
  }
  int hi = lane >> 4;
#pragma unroll
  for (int j = 0; j < 4; ++j) {
    int n = n0 + 16 * j + (lane & 15);
    float bv = bias ? bias[n] : 0.0f;
#pragma unroll
    for (int r = 0; r < 8; ++r) {
      C[(size_t)(m0 + r + 8 * hi) * ldc + n] = (OutT)(acc0[j][r] + bv);
      C[(size_t)(m0 + 16 + r + 8 * hi) * ldc + n] = (OutT)(acc1[j][r] + bv);
    }
  }
}

// ---------------- LayerNorm (one 512-wide row per block) -------------------

__global__ void layernorm_kernel(const float* __restrict__ in,
                                 const float* __restrict__ res,
                                 const float* __restrict__ g,
                                 const float* __restrict__ bta, float eps,
                                 float* __restrict__ out_f32,
                                 _Float16* __restrict__ out_f16) {
  int row = blockIdx.x, t = threadIdx.x;
  float v = in[(size_t)row * HH + t];
  if (res) v += res[(size_t)row * HH + t];
  __shared__ float sdata[HH];
  __shared__ float smu, svar;
  sdata[t] = v;
  __syncthreads();
  for (int s = HH / 2; s > 0; s >>= 1) {
    if (t < s) sdata[t] += sdata[t + s];
    __syncthreads();
  }
  if (t == 0) smu = sdata[0] * (1.0f / HH);
  __syncthreads();
  float mu = smu, d = v - mu;
  sdata[t] = d * d;
  __syncthreads();
  for (int s = HH / 2; s > 0; s >>= 1) {
    if (t < s) sdata[t] += sdata[t + s];
    __syncthreads();
  }
  if (t == 0) svar = sdata[0] * (1.0f / HH);
  __syncthreads();
  float y = d * rsqrtf(svar + eps) * g[t] + bta[t];
  out_f32[(size_t)row * HH + t] = y;
  if (out_f16) out_f16[(size_t)row * HH + t] = (_Float16)y;
}

// ---------------- LSTM recurrence: one block per (batch, direction) -------

__global__ void lstm_kernel(const float* __restrict__ pre_f,
                            const float* __restrict__ pre_b,
                            const float* __restrict__ whh_f,
                            const float* __restrict__ whh_b,
                            float* __restrict__ hf, float* __restrict__ hb) {
  int b = blockIdx.x & 31;
  int dir = blockIdx.x >> 5;
  const float* pre = dir ? pre_b : pre_f;
  const float* whh = dir ? whh_b : whh_f;
  float* hout = dir ? hb : hf;
  __shared__ float hs[HC];
  int t = threadIdx.x;
  hs[t] = 0.0f;
  float c = 0.0f;
  __syncthreads();
  const float4* w0 = (const float4*)(whh + (size_t)t * HC);
  const float4* w1 = (const float4*)(whh + (size_t)(HC + t) * HC);
  const float4* w2 = (const float4*)(whh + (size_t)(2 * HC + t) * HC);
  const float4* w3 = (const float4*)(whh + (size_t)(3 * HC + t) * HC);
  for (int s = 0; s < LL; ++s) {
    int l = dir ? (LL - 1 - s) : s;
    const float* prow = pre + ((size_t)(b * LL + l)) * (4 * HC);
    float ai = prow[t], af = prow[HC + t], ag = prow[2 * HC + t],
          ao = prow[3 * HC + t];
    const float4* h4 = (const float4*)hs;
    for (int k = 0; k < HC / 4; ++k) {
      float4 hv = h4[k];
      float4 a0 = w0[k], a1 = w1[k], a2 = w2[k], a3 = w3[k];
      ai += hv.x * a0.x + hv.y * a0.y + hv.z * a0.z + hv.w * a0.w;
      af += hv.x * a1.x + hv.y * a1.y + hv.z * a1.z + hv.w * a1.w;
      ag += hv.x * a2.x + hv.y * a2.y + hv.z * a2.z + hv.w * a2.w;
      ao += hv.x * a3.x + hv.y * a3.y + hv.z * a3.z + hv.w * a3.w;
    }
    float gi = 1.0f / (1.0f + __expf(-ai));
    float gf = 1.0f / (1.0f + __expf(-af));
    float gg = tanhf(ag);
    float go = 1.0f / (1.0f + __expf(-ao));
    c = gf * c + gi * gg;
    float h = go * tanhf(c);
    __syncthreads();
    hs[t] = h;
    hout[((size_t)(b * LL + l)) * HC + t] = h;
    __syncthreads();
  }
}

// ---------------- fused flash attention with rel-pos band WMMA -------------
// Grid (L/64, NH, B); 4 waves/block, each wave owns a 16-row l-tile.
// scores = (q.k^T + q.rel_emb[dist]^T)/sqrt(HD) + mask, online softmax, p.v.

__global__ void flash_attn_kernel(const _Float16* __restrict__ qh,
                                  const _Float16* __restrict__ kh,
                                  const _Float16* __restrict__ vh,
                                  const _Float16* __restrict__ relh,
                                  const float* __restrict__ span_mask,
                                  float* __restrict__ ctx) {
  __shared__ float Tband[4][16][48];   // skewed rel-pos scores per wave
  __shared__ _Float16 pbuf[4][16][32]; // prob tile -> A fragment staging
  int w = threadIdx.x >> 5, lane = threadIdx.x & 31;
  int hi = lane >> 4, col = lane & 15;
  int l0 = (blockIdx.x * 4 + w) * 16;
  int h = blockIdx.y, b = blockIdx.z;
  const _Float16* qb = qh + ((size_t)b * LL) * HH + h * HD;
  const _Float16* kb = kh + ((size_t)b * LL) * HH + h * HD;
  const _Float16* vb = vh + ((size_t)b * LL) * HH + h * HD;
  const float* mrow = span_mask + (size_t)b * LL;

  v16h qf = load_a_frag(qb, HH, l0, 0); // HD == 32: exactly one K-step
  v8f o0 = {}, o1 = {};
  float rowm[8], rowl[8];
#pragma unroll
  for (int r = 0; r < 8; ++r) { rowm[r] = -1e30f; rowl[r] = 0.0f; }

  for (int m0 = 0; m0 < LL; m0 += 32) {
    v8f z = {};
    v8f s0 = wmma_f16(qf, load_b_frag_nk(kb, HH, m0, 0), z);
    v8f s1 = wmma_f16(qf, load_b_frag_nk(kb, HH, m0 + 16, 0), z);
    // dist = l - m + 511 spans [rbase, rbase+46]; relh row 1023 is zero pad.
    int rbase = l0 - m0 + 480;
    v8f t0 = wmma_f16(qf, load_b_frag_nk(relh, HD, rbase, 0), z);
    v8f t1 = wmma_f16(qf, load_b_frag_nk(relh, HD, rbase + 16, 0), z);
    v8f t2 = wmma_f16(qf, load_b_frag_nk(relh, HD, rbase + 32, 0), z);
#pragma unroll
    for (int r = 0; r < 8; ++r) {
      int row = r + 8 * hi;
      Tband[w][row][col] = t0[r];
      Tband[w][row][16 + col] = t1[r];
      Tband[w][row][32 + col] = t2[r];
    }
    __syncthreads();
    float sv[16];
    float msk0 = (1.0f - mrow[m0 + col]) * -10000.0f;
    float msk1 = (1.0f - mrow[m0 + 16 + col]) * -10000.0f;
#pragma unroll
    for (int r = 0; r < 8; ++r) {
      int row = r + 8 * hi;
      sv[r] = (s0[r] + Tband[w][row][row - col + 31]) * 0.17677669529663689f +
              msk0;
      sv[r + 8] = (s1[r] + Tband[w][row][row - col + 15]) *
                      0.17677669529663689f + msk1;
    }
    __syncthreads();
#pragma unroll
    for (int r = 0; r < 8; ++r) {
      float mx = fmaxf(sv[r], sv[r + 8]);
      mx = fmaxf(mx, __shfl_xor(mx, 1));
      mx = fmaxf(mx, __shfl_xor(mx, 2));
      mx = fmaxf(mx, __shfl_xor(mx, 4));
      mx = fmaxf(mx, __shfl_xor(mx, 8));
      float mnew = fmaxf(rowm[r], mx);
      float p0 = __expf(sv[r] - mnew);
      float p1 = __expf(sv[r + 8] - mnew);
      float ps = p0 + p1;
      ps += __shfl_xor(ps, 1);
      ps += __shfl_xor(ps, 2);
      ps += __shfl_xor(ps, 4);
      ps += __shfl_xor(ps, 8);
      float esc = __expf(rowm[r] - mnew);
      rowl[r] = rowl[r] * esc + ps;
      rowm[r] = mnew;
      o0[r] *= esc;
      o1[r] *= esc;
      int row = r + 8 * hi;
      pbuf[w][row][col] = (_Float16)p0;
      pbuf[w][row][16 + col] = (_Float16)p1;
    }
    __syncthreads();
    v16h pf = load_a_frag(&pbuf[w][0][0], 32, 0, 0);
    o0 = wmma_f16(pf, load_b_frag_kn(vb, HH, m0, 0), o0);
    o1 = wmma_f16(pf, load_b_frag_kn(vb, HH, m0, 16), o1);
    __syncthreads();
  }
#pragma unroll
  for (int r = 0; r < 8; ++r) {
    int row = l0 + r + 8 * hi;
    float inv = 1.0f / rowl[r];
    size_t base = ((size_t)b * LL + row) * HH + h * HD;
    ctx[base + col] = o0[r] * inv;
    ctx[base + 16 + col] = o1[r] * inv;
  }
}

// ---------------- classifier + masked NLL loss ----------------------------

__global__ void zero_kernel(float* p, int n) {
  int i = blockIdx.x * blockDim.x + threadIdx.x;
  if (i < n) p[i] = 0.0f;
}

__global__ void clf_loss_kernel(const float* __restrict__ hf,
                                const float* __restrict__ hb,
                                const float* __restrict__ attnln,
                                const float* __restrict__ clf_w,
                                const float* __restrict__ clf_b,
                                const int* __restrict__ labels,
                                float* __restrict__ out_logits,
                                float* __restrict__ lossbuf) {
  int row = blockIdx.x * blockDim.x + threadIdx.x;
  if (row >= BB * LL) return;
  const float* f1 = hf + (size_t)row * HC;
  const float* f2 = hb + (size_t)row * HC;
  const float* f3 = attnln + (size_t)row * HH;
  float lg[NLB];
#pragma unroll
  for (int n = 0; n < NLB; ++n) {
    const float* wr = clf_w + (size_t)n * (2 * HH);
    float a = clf_b[n];
    for (int k = 0; k < HC; ++k) a += f1[k] * wr[k];
    for (int k = 0; k < HC; ++k) a += f2[k] * wr[HC + k];
    for (int k = 0; k < HH; ++k) a += f3[k] * wr[HH + k];
    lg[n] = a;
    out_logits[(size_t)row * NLB + n] = a;
  }
  int lab = labels[row];
  bool valid = lab >= 0;
  int li = valid ? lab : 0;
  float mx = fmaxf(lg[0], fmaxf(lg[1], lg[2]));
  float se = __expf(lg[0] - mx) + __expf(lg[1] - mx) + __expf(lg[2] - mx);
  float logp = lg[li] - mx - logf(se);
  if (valid) {
    atomicAdd(&lossbuf[0], -logp);
    atomicAdd(&lossbuf[1], 1.0f);
  }
}

__global__ void finalize_loss_kernel(const float* lossbuf, float* out_loss) {
  out_loss[0] = lossbuf[0] / fmaxf(lossbuf[1], 1.0f);
}

// ---------------------------------------------------------------------------

extern "C" void kernel_launch(void* const* d_in, const int* in_sizes, int n_in,
                              void* d_out, int out_size, void* d_ws,
                              size_t ws_size, hipStream_t stream) {
  const float* meta = (const float*)d_in[0];
  const float* span_mask = (const float*)d_in[2];
  const int* labels = (const int*)d_in[4];
  const float* proj_w = (const float*)d_in[5];
  const float* proj_b = (const float*)d_in[6];
  const float* ln_g = (const float*)d_in[7];
  const float* ln_b = (const float*)d_in[8];
  const float* wih_f = (const float*)d_in[9];
  const float* whh_f = (const float*)d_in[10];
  const float* lb_f = (const float*)d_in[11];
  const float* wih_b = (const float*)d_in[12];
  const float* whh_b = (const float*)d_in[13];
  const float* lb_b = (const float*)d_in[14];
  const float* q_w = (const float*)d_in[15];
  const float* q_b = (const float*)d_in[16];
  const float* k_w = (const float*)d_in[17];
  const float* k_b = (const float*)d_in[18];
  const float* v_w = (const float*)d_in[19];
  const float* v_b = (const float*)d_in[20];
  const float* rel_emb = (const float*)d_in[21];
  const float* ao_w = (const float*)d_in[22];
  const float* ao_b = (const float*)d_in[23];
  const float* aln_g = (const float*)d_in[24];
  const float* aln_b = (const float*)d_in[25];
  const float* clf_w = (const float*)d_in[26];
  const float* clf_b = (const float*)d_in[27];
  float* out_logits = (float*)d_out;
  float* out_loss = out_logits + (size_t)BB * LL * NLB;

  char* p = (char*)d_ws;
  auto alloc = [&](size_t bytes) -> void* {
    void* r = (void*)p;
    p += (bytes + 255) & ~(size_t)255;
    return r;
  };
  const size_t ROWS = (size_t)BB * LL; // 16384
  _Float16* metah = (_Float16*)alloc(ROWS * FF * 2);
  _Float16* projwh = (_Float16*)alloc((size_t)HH * FF * 2);
  _Float16* wihfh = (_Float16*)alloc((size_t)4 * HC * HH * 2);
  _Float16* wihbh = (_Float16*)alloc((size_t)4 * HC * HH * 2);
  _Float16* qwh = (_Float16*)alloc((size_t)HH * HH * 2);
  _Float16* kwh = (_Float16*)alloc((size_t)HH * HH * 2);
  _Float16* vwh = (_Float16*)alloc((size_t)HH * HH * 2);
  _Float16* aowh = (_Float16*)alloc((size_t)HH * HH * 2);
  _Float16* relh = (_Float16*)alloc((size_t)1024 * HD * 2); // padded row 1023
  float* xraw = (float*)alloc(ROWS * HH * 4);
  float* x = (float*)alloc(ROWS * HH * 4);
  _Float16* xh = (_Float16*)alloc(ROWS * HH * 2);
  float* pre_f = (float*)alloc(ROWS * 4 * HC * 4);
  float* pre_b = (float*)alloc(ROWS * 4 * HC * 4);
  float* hf = (float*)alloc(ROWS * HC * 4);
  float* hb = (float*)alloc(ROWS * HC * 4);
  _Float16* qh = (_Float16*)alloc(ROWS * HH * 2);
  _Float16* kh = (_Float16*)alloc(ROWS * HH * 2);
  _Float16* vh = (_Float16*)alloc(ROWS * HH * 2);
  float* ctx = (float*)alloc(ROWS * HH * 4);
  _Float16* ctxh = (_Float16*)alloc(ROWS * HH * 2);
  float* attnraw = (float*)alloc(ROWS * HH * 4);
  float* attnln = (float*)alloc(ROWS * HH * 4);
  float* lossbuf = (float*)alloc(256);

  auto cvt = [&](const float* s, _Float16* d, int n, int srcn) {
    cvt_f16_kernel<<<(n + 255) / 256, 256, 0, stream>>>(s, d, n, srcn);
  };
  // f32 -> f16 conversions (weights + activations as needed)
  cvt(meta, metah, ROWS * FF, ROWS * FF);
  cvt(proj_w, projwh, HH * FF, HH * FF);
  cvt(wih_f, wihfh, 4 * HC * HH, 4 * HC * HH);
  cvt(wih_b, wihbh, 4 * HC * HH, 4 * HC * HH);
  cvt(q_w, qwh, HH * HH, HH * HH);
  cvt(k_w, kwh, HH * HH, HH * HH);
  cvt(v_w, vwh, HH * HH, HH * HH);
  cvt(ao_w, aowh, HH * HH, HH * HH);
  cvt(rel_emb, relh, 1024 * HD, 1023 * HD); // zero-pad row 1023

  // x = LN(meta @ proj_w^T + b)
  gemm_wmma_kernel<float><<<dim3(HH / 64, ROWS / 128), 128, 0, stream>>>(
      metah, projwh, proj_b, xraw, (int)ROWS, HH, FF, FF, FF, HH);
  layernorm_kernel<<<(int)ROWS, HH, 0, stream>>>(xraw, nullptr, ln_g, ln_b,
                                                 1e-7f, x, xh);
  // LSTM pre-gates (both directions), then the sequential recurrence
  gemm_wmma_kernel<float><<<dim3(4 * HC / 64, ROWS / 128), 128, 0, stream>>>(
      xh, wihfh, lb_f, pre_f, (int)ROWS, 4 * HC, HH, HH, HH, 4 * HC);
  gemm_wmma_kernel<float><<<dim3(4 * HC / 64, ROWS / 128), 128, 0, stream>>>(
      xh, wihbh, lb_b, pre_b, (int)ROWS, 4 * HC, HH, HH, HH, 4 * HC);
  lstm_kernel<<<64, HC, 0, stream>>>(pre_f, pre_b, whh_f, whh_b, hf, hb);

  // Q, K, V projections straight to f16
  gemm_wmma_kernel<_Float16><<<dim3(HH / 64, ROWS / 128), 128, 0, stream>>>(
      xh, qwh, q_b, qh, (int)ROWS, HH, HH, HH, HH, HH);
  gemm_wmma_kernel<_Float16><<<dim3(HH / 64, ROWS / 128), 128, 0, stream>>>(
      xh, kwh, k_b, kh, (int)ROWS, HH, HH, HH, HH, HH);
  gemm_wmma_kernel<_Float16><<<dim3(HH / 64, ROWS / 128), 128, 0, stream>>>(
      xh, vwh, v_b, vh, (int)ROWS, HH, HH, HH, HH, HH);

  // fused attention (scores never materialized)
  flash_attn_kernel<<<dim3(LL / 64, NH, BB), 128, 0, stream>>>(
      qh, kh, vh, relh, span_mask, ctx);

  // attn_out = LN(ctx @ W^T + b + x)
  cvt(ctx, ctxh, ROWS * HH, ROWS * HH);
  gemm_wmma_kernel<float><<<dim3(HH / 64, ROWS / 128), 128, 0, stream>>>(
      ctxh, aowh, ao_b, attnraw, (int)ROWS, HH, HH, HH, HH, HH);
  layernorm_kernel<<<(int)ROWS, HH, 0, stream>>>(attnraw, x, aln_g, aln_b,
                                                 1e-12f, attnln, nullptr);

  // classifier + loss
  zero_kernel<<<1, 32, 0, stream>>>(lossbuf, 2);
  clf_loss_kernel<<<(int)(ROWS / 256), 256, 0, stream>>>(
      hf, hb, attnln, clf_w, clf_b, labels, out_logits, lossbuf);
  finalize_loss_kernel<<<1, 1, 0, stream>>>(lossbuf, out_loss);
}
